// MultiheadAttention_22789096473378
// MI455X (gfx1250) — matrix operand order repaired and checked by reading, not compiled, not run on hardware
//
#include <hip/hip_runtime.h>
#include <hip/hip_bf16.h>

typedef _Float16 h8   __attribute__((ext_vector_type(8)));
typedef _Float16 v16h __attribute__((ext_vector_type(16)));
typedef float    v8f  __attribute__((ext_vector_type(8)));
typedef float    f4   __attribute__((ext_vector_type(4)));

#define DEV __device__ __forceinline__

constexpr int D   = 1024;   // embed dim
constexpr int S   = 2048;   // sequence
constexpr int H   = 16;     // heads
constexpr int DH  = 64;     // head dim
constexpr int BSZ = 4;      // batch
constexpr int MT  = BSZ * S;                  // 8192 total rows
constexpr int BM  = 128, BN = 256, BK = 32;   // GEMM block tile
constexpr int LP  = BK + 8;                   // padded LDS stride (40 halves)

// ---- helpers -------------------------------------------------------------

DEV h8 cvt8(const float* p) {           // 8 x f32 -> 8 x f16
  f4 a = *(const f4*)p;
  f4 b = *(const f4*)(p + 4);
  h8 r;
  r[0]=(_Float16)a[0]; r[1]=(_Float16)a[1]; r[2]=(_Float16)a[2]; r[3]=(_Float16)a[3];
  r[4]=(_Float16)b[0]; r[5]=(_Float16)b[1]; r[6]=(_Float16)b[2]; r[7]=(_Float16)b[3];
  return r;
}
DEV h8 cvt8(const _Float16* p) { return *(const h8*)p; }

DEV v16h cat16(h8 lo, h8 hi) {
  return __builtin_shufflevector(lo, hi, 0,1,2,3,4,5,6,7,8,9,10,11,12,13,14,15);
}

DEV v8f wmma_f16(v16h a, v16h b, v8f c) {
  // D = A(16x32 f16) x B(32x16 f16) + C(16x16 f32)
  return __builtin_amdgcn_wmma_f32_16x16x32_f16(
      /*neg_a=*/false, a, /*neg_b=*/false, b,
      /*c_mod=*/(short)0, c, /*reuse_a=*/false, /*reuse_b=*/false);
}

DEV float sigmoid_scaled(float s) {
  // sigmoid(s/8) = 1 / (1 + exp2(s * (-0.125 * log2 e)))
  // one v_mul + v_exp_f32 + v_add + v_rcp_f32
  constexpr float C = -0.18033688011112042f;   // -0.125 * log2(e)
  return __builtin_amdgcn_rcpf(1.0f + __builtin_amdgcn_exp2f(s * C));
}

// async global -> LDS copy of 16 bytes per lane (ASYNCcnt-tracked, no VGPR data)
DEV void async_copy_b128(unsigned lds_off, const void* gaddr) {
  asm volatile("global_load_async_to_lds_b128 %0, %1, off"
               :: "v"(lds_off), "v"(gaddr) : "memory");
}
DEV void async_copy_b128_o16(unsigned lds_off, const void* gaddr) {
  // INST_OFFSET applies to both LDS and global address (ISA 08 §4.4)
  asm volatile("global_load_async_to_lds_b128 %0, %1, off offset:16"
               :: "v"(lds_off), "v"(gaddr) : "memory");
}
DEV void wait_async0() {
  asm volatile("s_wait_asynccnt 0x0" ::: "memory");
}

// ---- projection / output GEMM -------------------------------------------
// Y[row, col] = sum_d A[row, d] * W[col, d] + bias[col]
// MODE 0: write f16 to [B,H,S,Dh]      (Q, K projections)
// MODE 1: write f16 to [B,H,Dh,S]      (V projection, transposed for B-frags)
// MODE 2: write f32 to [B,S,D]         (final output projection)
template <typename AT, int MODE>
__global__ __launch_bounds__(256)
void gemm_wmma(const AT* __restrict__ A, const float* __restrict__ W,
               const float* __restrict__ bias, void* __restrict__ Out)
{
  __shared__ _Float16 As[2][BM][LP];   // double-buffered: 2 x 10 KB
  __shared__ _Float16 Bs[2][BN][LP];   // double-buffered: 2 x 20 KB

  const int tid  = threadIdx.x;
  const int bm0  = (blockIdx.x % (MT / BM)) * BM;
  const int bn0  = (blockIdx.x / (MT / BM)) * BN;

  const int lane = tid & 31;
  const int w    = tid >> 5;
  const int lr   = lane & 15;     // row/col selector within tile
  const int lh   = lane >> 4;     // half-wave selector
  const int wm   = (w >> 2) * 64; // wave M origin (2 wave-rows)
  const int wn   = (w & 3) * 64;  // wave N origin (4 wave-cols)

  const int arow = tid >> 1;             // A staging: 128 rows x 2 halves
  const int acol = (tid & 1) * 16;

  auto stage = [&](int buf, int kb) {
    const AT* ap = A + (size_t)(bm0 + arow) * D + kb + acol;
    *(h8*)&As[buf][arow][acol]     = cvt8(ap);
    *(h8*)&As[buf][arow][acol + 8] = cvt8(ap + 8);
    const float* wp = W + (size_t)(bn0 + tid) * D + kb;
#pragma unroll
    for (int u = 0; u < 4; ++u)
      *(h8*)&Bs[buf][tid][u * 8] = cvt8(wp + u * 8);
    if (kb + BK < D) {   // pull the tile after next toward the caches
      __builtin_prefetch((const void*)(ap + BK), 0, 0);
      __builtin_prefetch((const void*)(wp + BK), 0, 0);
    }
  };

  v8f acc[4][4];
#pragma unroll
  for (int nt = 0; nt < 4; ++nt) {
    const float bv = bias[bn0 + wn + nt * 16 + lr];
#pragma unroll
    for (int mt = 0; mt < 4; ++mt)
#pragma unroll
      for (int e = 0; e < 8; ++e) acc[mt][nt][e] = bv;
  }

  stage(0, 0);
  int cur = 0;
  for (int kb = 0; kb < D; kb += BK, cur ^= 1) {
    __syncthreads();                          // stage(cur) visible; prev reads done
    if (kb + BK < D) stage(cur ^ 1, kb + BK); // overlap next tile with compute

    v16h af[4], bf[4];
#pragma unroll
    for (int mt = 0; mt < 4; ++mt) {   // A-frag: m=lr, k = 16*(i/8)+8*lh+i%8
      const int m = wm + mt * 16 + lr;
      af[mt] = cat16(*(const h8*)&As[cur][m][8 * lh],
                     *(const h8*)&As[cur][m][16 + 8 * lh]);
    }
#pragma unroll
    for (int nt = 0; nt < 4; ++nt) {   // B-frag: n=lr, k = 16*(i/8)+8*lh+i%8
      const int n = wn + nt * 16 + lr;
      bf[nt] = cat16(*(const h8*)&Bs[cur][n][8 * lh],
                     *(const h8*)&Bs[cur][n][16 + 8 * lh]);
    }
#pragma unroll
    for (int mt = 0; mt < 4; ++mt)
#pragma unroll
      for (int nt = 0; nt < 4; ++nt)
        acc[mt][nt] = wmma_f16(af[mt], bf[nt], acc[mt][nt]);
  }

  // epilogue: C-frag lane layout M = r + 8*lh, N = lr
#pragma unroll
  for (int mt = 0; mt < 4; ++mt) {
#pragma unroll
    for (int nt = 0; nt < 4; ++nt) {
      const int col = bn0 + wn + nt * 16 + lr;
#pragma unroll
      for (int r = 0; r < 8; ++r) {
        const int row = bm0 + wm + mt * 16 + r + 8 * lh;
        const float v = acc[mt][nt][r];
        if constexpr (MODE == 2) {
          ((float*)Out)[(size_t)row * D + col] = v;
        } else {
          const int b  = row >> 11;        // row / S
          const int s  = row & (S - 1);
          const int hh = col >> 6;         // col / DH
          const int dh = col & (DH - 1);
          if constexpr (MODE == 0)
            ((_Float16*)Out)[((((size_t)b * H + hh) * S + s) << 6) + dh] = (_Float16)v;
          else
            ((_Float16*)Out)[((((size_t)b * H + hh) * DH + dh) << 11) + s] = (_Float16)v;
        }
      }
    }
  }
}

// ---- sigmoid attention ----------------------------------------------------
// out[q, dh] = sum_k sigmoid(q.k / 8) * v[k, dh], per (b, h)
// 4 waves / block, each wave owns 16 queries x 64 dh; K/V tiles staged once
// per block via async global->LDS DMA (shared by the 4 waves).
__global__ __launch_bounds__(128)
void attn_sigmoid(const _Float16* __restrict__ qp,   // [B,H,S,DH]
                  const _Float16* __restrict__ kp,   // [B,H,S,DH]
                  const _Float16* __restrict__ vT,   // [B,H,DH,S]
                  _Float16* __restrict__ ao)         // [B,S,D]
{
  __shared__ _Float16 Ks[32][DH + 8];  // 32 keys x 64 dh   (staged)
  __shared__ _Float16 Vs[DH][BK + 8];  // 64 dh  x 32 keys  (staged, transposed)
  __shared__ _Float16 Ps[4][16][BK + 8];  // per-wave 16x32 sigmoid tile

  const int tid  = threadIdx.x;
  const int lane = tid & 31;
  const int w    = tid >> 5;
  const int lr   = lane & 15;
  const int lh   = lane >> 4;

  const int qblk = blockIdx.x & (S / 64 - 1);  // 32 query blocks of 64
  const int bh   = blockIdx.x >> 5;            // b*H + h
  const int b    = bh >> 4;
  const int h    = bh & (H - 1);
  const int q0   = qblk * 64 + w * 16;

  // K staging map: 32 rows x 64 cols, 128 threads x 32 B
  const int krow = tid >> 2, kcol = (tid & 3) * 16;
  // V staging map: 64 rows x 32 cols, 128 threads x 32 B
  const int vrow_s = tid >> 1, vcol_s = (tid & 1) * 16;

  // wave-relative LDS offsets (generic-pointer low 32 bits == LDS address)
  const unsigned ldsK = (unsigned)(uintptr_t)&Ks[krow][kcol];
  const unsigned ldsV = (unsigned)(uintptr_t)&Vs[vrow_s][vcol_s];

  // Q fragments for this wave's 16 rows (two K=32 chunks over DH=64)
  const _Float16* qrow = qp + ((size_t)bh * S + q0 + lr) * DH;
  v16h qf[2];
#pragma unroll
  for (int c = 0; c < 2; ++c)
    qf[c] = cat16(*(const h8*)(qrow + c * 32 + 8 * lh),
                  *(const h8*)(qrow + c * 32 + 16 + 8 * lh));

  v8f o[4];
#pragma unroll
  for (int nt = 0; nt < 4; ++nt)
#pragma unroll
    for (int e = 0; e < 8; ++e) o[nt][e] = 0.0f;

  for (int kbase = 0; kbase < S; kbase += 32) {
    __syncthreads();   // previous iteration's frag reads complete
    {  // async-stage K tile Ks[key][dh] and V tile Vs[dh][key]
      const _Float16* kg = kp + ((size_t)bh * S + kbase + krow) * DH + kcol;
      const _Float16* vg = vT + ((size_t)bh * DH + vrow_s) * S + kbase + vcol_s;
      async_copy_b128(ldsK, kg);
      async_copy_b128_o16(ldsK, kg);
      async_copy_b128(ldsV, vg);
      async_copy_b128_o16(ldsV, vg);
      wait_async0();
    }
    __syncthreads();

    // ---- scores S = q . k^T over DH, for 16q x 32keys ----
    v8f sf[2];
#pragma unroll
    for (int nt = 0; nt < 2; ++nt)
#pragma unroll
      for (int e = 0; e < 8; ++e) sf[nt][e] = 0.0f;

#pragma unroll
    for (int nt = 0; nt < 2; ++nt) {   // B-frag: n(key)=nt*16+lr, dh in table order
#pragma unroll
      for (int c = 0; c < 2; ++c) {
        v16h bf = cat16(*(const h8*)&Ks[nt * 16 + lr][c * 32 + 8 * lh],
                        *(const h8*)&Ks[nt * 16 + lr][c * 32 + 16 + 8 * lh]);
        sf[nt] = wmma_f16(qf[c], bf, sf[nt]);
      }
    }

    // ---- sigmoid(s/8) -> P tile in wave-private LDS (C-layout -> A-layout)
#pragma unroll
    for (int nt = 0; nt < 2; ++nt)
#pragma unroll
      for (int r = 0; r < 8; ++r) {
        const float p = sigmoid_scaled(sf[nt][r]);
        Ps[w][r + 8 * lh][nt * 16 + lr] = (_Float16)p;
      }
    // DS ops are in-order within a wave: read back as A-fragment
    v16h pf = cat16(*(const h8*)&Ps[w][lr][8 * lh],
                    *(const h8*)&Ps[w][lr][16 + 8 * lh]);

    // ---- out += P(16x32) x V(32keys x 64dh) from staged Vs ----
#pragma unroll
    for (int nt = 0; nt < 4; ++nt) {   // B-frag: n(dh)=nt*16+lr, keys in table order
      v16h vf = cat16(*(const h8*)&Vs[nt * 16 + lr][8 * lh],
                      *(const h8*)&Vs[nt * 16 + lr][16 + 8 * lh]);
      o[nt] = wmma_f16(pf, vf, o[nt]);
    }
  }

  // epilogue -> ao[B,S,D] f16, interleaved heads (ready for Wo GEMM)
#pragma unroll
  for (int nt = 0; nt < 4; ++nt)
#pragma unroll
    for (int r = 0; r < 8; ++r) {
      const int qg = q0 + r + 8 * lh;
      ao[((size_t)b * S + qg) * D + h * DH + nt * 16 + lr] = (_Float16)o[nt][r];
    }
}

// ---- launcher -------------------------------------------------------------

extern "C" void kernel_launch(void* const* d_in, const int* in_sizes, int n_in,
                              void* d_out, int out_size, void* d_ws, size_t ws_size,
                              hipStream_t stream) {
  (void)in_sizes; (void)n_in; (void)out_size; (void)ws_size;
  const float* query = (const float*)d_in[0];
  const float* key   = (const float*)d_in[1];
  const float* value = (const float*)d_in[2];
  const float* Wq = (const float*)d_in[3];
  const float* bq = (const float*)d_in[4];
  const float* Wk = (const float*)d_in[5];
  const float* bk = (const float*)d_in[6];
  const float* Wv = (const float*)d_in[7];
  const float* bv = (const float*)d_in[8];
  const float* Wo = (const float*)d_in[9];
  const float* bo = (const float*)d_in[10];

  // workspace: 4 x 16 MiB f16 tensors = 64 MiB
  _Float16* qp = (_Float16*)d_ws;               // [B,H,S,DH]
  _Float16* kp = qp + (size_t)MT * D;           // [B,H,S,DH]
  _Float16* vT = kp + (size_t)MT * D;           // [B,H,DH,S]
  _Float16* ao = vT + (size_t)MT * D;           // [B,S,D]

  const int gemmGrid = (MT / BM) * (D / BN);    // 64 * 4 = 256 blocks
  gemm_wmma<float, 0><<<gemmGrid, 256, 0, stream>>>(query, Wq, bq, (void*)qp);
  gemm_wmma<float, 0><<<gemmGrid, 256, 0, stream>>>(key,   Wk, bk, (void*)kp);
  gemm_wmma<float, 1><<<gemmGrid, 256, 0, stream>>>(value, Wv, bv, (void*)vT);
  attn_sigmoid<<<BSZ * H * (S / 64), 128, 0, stream>>>(qp, kp, vT, ao);
  gemm_wmma<_Float16, 2><<<gemmGrid, 256, 0, stream>>>(ao, Wo, bo, d_out);
}
